// InverseDistanceAttention_79010218377514
// MI455X (gfx1250) — compile-verified, matrix operand order checked by reading, not run on hardware
//
#include <hip/hip_runtime.h>
#include <cstdint>

// ---------------- types ----------------
typedef __attribute__((ext_vector_type(16))) _Float16      v16h;
typedef __attribute__((ext_vector_type(8)))  float         v8f;
typedef __attribute__((ext_vector_type(4)))  unsigned int  u32x4;
typedef __attribute__((ext_vector_type(8)))  int           i32x8;
typedef __attribute__((ext_vector_type(4)))  int           i32x4;

#define NF 256   // feature dim F
#define NL 64    // stations L
#define NS 168   // sequence S
#define NB 32    // batch B

// ------------------------------------------------------------------
// Prologue: repack W1 (f32, [g=256][f=256]) into f16 B-matrix fragments.
// GEMM is h[l,g] = sum_f A[l,f] * W1[g,f]  ->  B_mat[k=f, n=g] = W1[g,f].
// 16-bit B fragment layout (wave32, 16x16x32): lane = 16*lg + n,
// element e -> K = e + 8*lg + 8*(e>=8).  One fragment = 32 contiguous bytes.
// Buffer layout: [(kt*16+nt)*32 + lane][16 halves]; kt in 0..7, nt in 0..15.
// ------------------------------------------------------------------
__global__ void idw_cvt_w1(const float* __restrict__ W1,
                           _Float16* __restrict__ out) {
  int idx  = blockIdx.x * blockDim.x + threadIdx.x;   // 0..4095
  int lane = idx & 31;
  int tile = idx >> 5;                                // kt*16 + nt
  int kt = tile >> 4, nt = tile & 15;
  int lg = lane >> 4, n = lane & 15;
  const float* src = W1 + (size_t)(nt * 16 + n) * NF + kt * 32 + 8 * lg;
  _Float16* dst = out + (size_t)idx * 16;
#pragma unroll
  for (int e = 0; e < 8; ++e) dst[e] = (_Float16)src[e];          // K = 8*lg + e
#pragma unroll
  for (int e = 0; e < 8; ++e) dst[8 + e] = (_Float16)src[16 + e]; // K = 16 + 8*lg + e
}

// ------------------------------------------------------------------
// Fused kernel: one workgroup per (b,s).
//   TDM-load 64x256 f32 tile -> LDS; WMMA GEMM vs W1; fused
//   bias+ReLU+dot(w2)+ReLU; inverse-distance softmax over L; weighted sum.
// ------------------------------------------------------------------
__launch_bounds__(256)
__global__ void idw_attn_kernel(const float* __restrict__ feat,
                                const float* __restrict__ src_locs,
                                const float* __restrict__ tar_locs,
                                const _Float16* __restrict__ W1c,
                                const float* __restrict__ b1,
                                const float* __restrict__ w2,
                                const float* __restrict__ b2,
                                float* __restrict__ out) {
  __shared__ __align__(16) float Atile[NL * NF];  // 64 KB feature tile (f32)
  __shared__ float attnbuf[NL];
  __shared__ float expbuf[NL];

  const int tid = threadIdx.x;
  const int bs  = blockIdx.x;          // b*S + s
  const int b   = bs / NS;

  // zero the per-row attention accumulators (other waves, while wave 0 DMAs)
  if (tid >= 64 && tid < 128) attnbuf[tid - 64] = 0.0f;

  // Warm the repacked-W1 region (128 KB, shared by all WGs) while TDM runs.
  __builtin_prefetch((const void*)(W1c + (size_t)tid * 256), 0, 1);

  // ---- Phase 1: TDM DMA of the contiguous 64KB (L,F) tile into LDS ----
  if (tid < 32) {
    uint64_t gaddr = (uint64_t)(uintptr_t)(feat + (size_t)bs * (NL * NF));
    uint32_t laddr = (uint32_t)(uintptr_t)(&Atile[0]);
    u32x4 g0;
    g0.x = 1u;                                                   // count=1
    g0.y = laddr;                                                // lds_addr
    g0.z = (uint32_t)gaddr;                                      // global_addr lo
    g0.w = (uint32_t)((gaddr >> 32) & 0x1FFFFFFu) | (2u << 30);  // addr hi | type=2
    i32x8 g1;
    g1[0] = (2 << 16);              // data_size = 4 bytes, no multicast
    g1[1] = (int)(0x4000u << 16);   // tensor_dim0 = 16384 (bits 63:48)
    g1[2] = (1 << 16);              // tensor_dim0 hi=0 | tensor_dim1 = 1
    g1[3] = (int)(0x4000u << 16);   // tile_dim0 = 16384 (bits 127:112)
    g1[4] = 0;                      // tile_dim1/2 unused (1-D tile)
    g1[5] = 16384;                  // tensor_dim0_stride
    g1[6] = 0;
    g1[7] = 0;
    i32x4 g2 = {0, 0, 0, 0};
    i32x4 g3 = {0, 0, 0, 0};
    i32x8 g4 = {0, 0, 0, 0, 0, 0, 0, 0};   // unused trailing group (6-arg form)
    __builtin_amdgcn_tensor_load_to_lds(g0, g1, g2, g3, g4, 0);
    __builtin_amdgcn_s_wait_tensorcnt(0);
  }
  __syncthreads();

  // ---- Phase 2: WMMA GEMM + fused attention-score epilogue ----
  const int wave = tid >> 5;
  const int lane = tid & 31;
  const int lg   = lane >> 4;        // 16-lane group
  const int mn   = lane & 15;        // row (A) / column (B,D) within group
  const int mt   = wave & 3;         // M tile: rows mt*16 .. mt*16+15
  const int nt0  = (wave >> 2) << 3; // N tiles [nt0, nt0+8)

  // Preload the 8 bias / w2 values for this wave's columns (overlap w/ GEMM).
  float b1v[8], w2v[8];
#pragma unroll
  for (int t = 0; t < 8; ++t) {
    const int col = (nt0 + t) * 16 + mn;
    b1v[t] = b1[col];
    w2v[t] = w2[col];
  }

  // Build A fragments (f32 LDS -> f16), batched so ds_loads overlap.
  v16h afrag[8];
#pragma unroll
  for (int h2 = 0; h2 < 2; ++h2) {
    float4 t0[4], t1[4], t2[4], t3[4];
#pragma unroll
    for (int k4 = 0; k4 < 4; ++k4) {
      const float* ap = &Atile[(mt * 16 + mn) * NF + (h2 * 4 + k4) * 32 + 8 * lg];
      t0[k4] = *(const float4*)(ap);        // K = 8*lg + 0..3
      t1[k4] = *(const float4*)(ap + 4);    // K = 8*lg + 4..7
      t2[k4] = *(const float4*)(ap + 16);   // K = 16 + 8*lg + 0..3
      t3[k4] = *(const float4*)(ap + 20);   // K = 16 + 8*lg + 4..7
    }
#pragma unroll
    for (int k4 = 0; k4 < 4; ++k4) {
      v16h a;
      a[0]  = (_Float16)t0[k4].x; a[1]  = (_Float16)t0[k4].y;
      a[2]  = (_Float16)t0[k4].z; a[3]  = (_Float16)t0[k4].w;
      a[4]  = (_Float16)t1[k4].x; a[5]  = (_Float16)t1[k4].y;
      a[6]  = (_Float16)t1[k4].z; a[7]  = (_Float16)t1[k4].w;
      a[8]  = (_Float16)t2[k4].x; a[9]  = (_Float16)t2[k4].y;
      a[10] = (_Float16)t2[k4].z; a[11] = (_Float16)t2[k4].w;
      a[12] = (_Float16)t3[k4].x; a[13] = (_Float16)t3[k4].y;
      a[14] = (_Float16)t3[k4].z; a[15] = (_Float16)t3[k4].w;
      afrag[h2 * 4 + k4] = a;
    }
  }

  float partial[8];
#pragma unroll
  for (int r = 0; r < 8; ++r) partial[r] = 0.0f;

  // Fragment element offset: (kt*16 + nt)*32 + lane) * 16
  //                        = kt*8192 + nt*512 + lane*16
#pragma unroll
  for (int t = 0; t < 8; ++t) {
    const int nt = nt0 + t;
    const _Float16* bp = W1c + (size_t)(nt * 512 + lane * 16);
    v8f c = {};
    // 2-deep software pipeline: keep B loads 2 K-steps ahead of the WMMA.
    v16h bf0 = *(const v16h*)(bp);
    v16h bf1 = *(const v16h*)(bp + 8192);
#pragma unroll
    for (int kt = 0; kt < 8; ++kt) {
      const v16h bcur = (kt & 1) ? bf1 : bf0;
      if (kt + 2 < 8) {
        const v16h bnew = *(const v16h*)(bp + (size_t)(kt + 2) * 8192);
        if (kt & 1) bf1 = bnew; else bf0 = bnew;
      }
      c = __builtin_amdgcn_wmma_f32_16x16x32_f16(
          false, afrag[kt], false, bcur, (short)0, c, false, false);
    }
    // D layout: c[r] = H[mt*16 + r + 8*lg][nt*16 + mn]
#pragma unroll
    for (int r = 0; r < 8; ++r) {
      float h = c[r] + b1v[t];
      h = h > 0.0f ? h : 0.0f;           // relu of hidden
      partial[r] += h * w2v[t];          // dot with w2
    }
  }

  // Reduce the 16 column-lanes of each group, accumulate per-row score.
#pragma unroll
  for (int r = 0; r < 8; ++r) {
    float v = partial[r];
    v += __shfl_xor(v, 1);
    v += __shfl_xor(v, 2);
    v += __shfl_xor(v, 4);
    v += __shfl_xor(v, 8);
    if (mn == 0) atomicAdd(&attnbuf[mt * 16 + r + 8 * lg], v);
  }
  __syncthreads();

  // ---- Phase 3: inverse-distance weights + softmax over L=64 ----
  if (tid < NL) {
    const float sx = src_locs[((size_t)b * NL + tid) * 2 + 0];
    const float sy = src_locs[((size_t)b * NL + tid) * 2 + 1];
    const float tx = tar_locs[b * 2 + 0];
    const float ty = tar_locs[b * 2 + 1];
    const float dx = sx - tx, dy = sy - ty;
    const float dist = sqrtf(dx * dx + dy * dy);
    const float invd = 0.001f / (dist + 1e-8f);
    float attn = attnbuf[tid] + b2[0];
    attn = attn > 0.0f ? attn : 0.0f;    // relu of attention score
    attnbuf[tid] = attn * invd;          // softmax logits
  }
  __syncthreads();

  float mx = attnbuf[0];
#pragma unroll 8
  for (int l = 1; l < NL; ++l) mx = fmaxf(mx, attnbuf[l]);
  if (tid < NL) expbuf[tid] = __expf(attnbuf[tid] - mx);
  __syncthreads();

  float sum = 0.0f;
#pragma unroll 8
  for (int l = 0; l < NL; ++l) sum += expbuf[l];
  const float rs = 1.0f / sum;

  // ---- Phase 4: outputs[b,s,f] = sum_l scores[l] * features[l,f] (f32) ----
  float acc = 0.0f;
#pragma unroll 8
  for (int l = 0; l < NL; ++l) acc += expbuf[l] * Atile[l * NF + tid];
  out[(size_t)bs * NF + tid] = acc * rs;
}

// ------------------------------------------------------------------
extern "C" void kernel_launch(void* const* d_in, const int* in_sizes, int n_in,
                              void* d_out, int out_size, void* d_ws, size_t ws_size,
                              hipStream_t stream) {
  const float* feat = (const float*)d_in[0];   // (B,S,L,F)
  const float* srcl = (const float*)d_in[1];   // (B,L,2)
  const float* tarl = (const float*)d_in[2];   // (B,2)
  const float* W1   = (const float*)d_in[3];   // (F,F)
  const float* b1   = (const float*)d_in[4];   // (F)
  const float* w2   = (const float*)d_in[5];   // (F)
  const float* b2   = (const float*)d_in[6];   // (1)

  _Float16* W1c = (_Float16*)d_ws;             // 128 KB repacked f16 W1

  idw_cvt_w1<<<16, 256, 0, stream>>>(W1, W1c);
  idw_attn_kernel<<<NB * NS, 256, 0, stream>>>(feat, srcl, tarl, W1c,
                                               b1, w2, b2, (float*)d_out);
}
